// Model_74594991997068
// MI455X (gfx1250) — matrix operand order repaired and checked by reading, not compiled, not run on hardware
//
#include <hip/hip_runtime.h>
#include <hip/hip_bf16.h>

typedef __attribute__((ext_vector_type(16))) _Float16 v16h;
typedef __attribute__((ext_vector_type(8)))  float    v8f;

#define BN_EPS 1e-4f

// ---------------------------------------------- f32 -> f16 into padded layout
// in: [B,C,H,W] f32 ; out: [B,C,H+2,W+2] f16 interior at (1,1)
__global__ __launch_bounds__(256) void cvt_pad_f16(const float* __restrict__ in,
                                                   _Float16* __restrict__ out,
                                                   int C, int H, int W, long total) {
  long i = (long)blockIdx.x * blockDim.x + threadIdx.x;
  if (i >= total) return;
  int x = (int)(i % W); long t = i / W;
  int y = (int)(t % H); t /= H;
  int c = (int)(t % C);
  int b = (int)(t / C);
  out[(((long)b * C + c) * (H + 2) + y + 1) * (W + 2) + x + 1] = (_Float16)in[i];
}

// ---------------------------------------------------- zero the 1-px halo rim
// Y: [N, Hp, Wp]; total = N * (2*Wp + 2*(Hp-2))
__global__ __launch_bounds__(256) void zero_border(_Float16* __restrict__ Y,
                                                   int Hp, int Wp, long total) {
  long i = (long)blockIdx.x * blockDim.x + threadIdx.x;
  if (i >= total) return;
  const int per = 2 * Wp + 2 * (Hp - 2);
  const long plane = i / per;
  const int j = (int)(i % per);
  int y, x;
  if (j < Wp)           { y = 0;      x = j; }
  else if (j < 2 * Wp)  { y = Hp - 1; x = j - Wp; }
  else { int jj = j - 2 * Wp; y = 1 + (jj >> 1); x = (jj & 1) ? (Wp - 1) : 0; }
  Y[(plane * Hp + y) * Wp + x] = (_Float16)0.f;
}

// -------------------------------------------------- weight fragment prepack
// frag[((chunk*NT + t)*32 + lane)*16 + e]: K = chunk*32 + 8*(lane>>4) + (v<4 ? 2v+p : 16+2(v-4)+p),
// N = t*16 + (lane&15); zero-padded in K and N (this makes the A-gather's
// padded-K garbage harmless: it multiplies 0).
__global__ __launch_bounds__(256) void prep_wfrag(const float* __restrict__ Wt,
                                                  _Float16* __restrict__ frag,
                                                  int OC, int KTOT, int NT, int kmajor, int Ncols,
                                                  long total) {
  long i = (long)blockIdx.x * blockDim.x + threadIdx.x;
  if (i >= total) return;
  const int e    = (int)(i & 15);
  const int lane = (int)((i >> 4) & 31);
  const long ct  = i >> 9;
  const int t     = (int)(ct % NT);
  const int chunk = (int)(ct / NT);
  const int g = lane >> 4;
  const int v = e >> 1, p = e & 1;
  const int k = chunk * 32 + g * 8 + ((v < 4) ? (2 * v + p) : (16 + 2 * (v - 4) + p));
  const int n = t * 16 + (lane & 15);
  float w = 0.f;
  if (n < OC && k < KTOT)
    w = kmajor ? Wt[(long)k * Ncols + n]     // [K, N] (linear weight)
               : Wt[(long)n * KTOT + k];     // [OC, KTOT] (conv weight)
  frag[i] = (_Float16)w;
}

// ------------------------------------------- fused conv + BN + ReLU (WMMA)
// Halo-padded input [B,IC,H+2P,W+2P] -> unconditional A gather (no masks).
// One wave32 = 16 consecutive x pixels of one (b,oy) row; K fully unrolled.
template<int IC, int OC, int KH, int KW, int PAD, int H, int W, int OPAD>
__global__ __launch_bounds__(256) void conv_bnrelu_wmma(
    const _Float16* __restrict__ X, const _Float16* __restrict__ Wfrag,
    const float* __restrict__ BN, _Float16* __restrict__ Y)
{
  constexpr int B    = 512;
  constexpr int KK   = KH * KW;
  constexpr int KTOT = IC * KK;
  constexpr int CH   = (KTOT + 31) / 32;
  constexpr int NT   = (OC + 15) / 16;
  constexpr int HIN  = H + 2 * PAD, WIN = W + 2 * PAD;
  constexpr int OH   = H + 2 * PAD - KH + 1, OW = W + 2 * PAD - KW + 1;
  constexpr int OHP  = OH + 2 * OPAD, OWP = OW + 2 * OPAD;
  constexpr int tilesX = (OW + 15) / 16;
  constexpr long totalTiles = (long)B * OH * tilesX;

  const int lane = threadIdx.x & 31;
  const int wvq  = threadIdx.x >> 5;
  const int mrow = lane & 15;
  const int g    = lane >> 4;

  const long tileM  = (long)blockIdx.x * 8 + wvq;
  const bool wvalid = (tileM < totalTiles);
  const int tm   = wvalid ? (int)tileM : 0;
  const int xt   = tm % tilesX;
  const int rowI = tm / tilesX;
  const int oy   = rowI % OH;
  const int b    = rowI / OH;
  const int x0   = xt * 16;
  const int ox   = x0 + mrow;

  const int pixBase = (b * IC * HIN + oy) * WIN + ox;   // fits in i32 for all layers
  const v16h* __restrict__ fragv = (const v16h*)Wfrag;
  v8f acc[NT] = {};
  __builtin_prefetch(Wfrag, 0, 3);

#pragma unroll
  for (int chunk = 0; chunk < CH; ++chunk) {
    // ---- A fragment: fully unconditional gather (halo padding + zero weights)
    v16h a;
#pragma unroll
    for (int e = 0; e < 16; ++e) {
      const int v  = e >> 1, p = e & 1;
      const int k0 = chunk * 32 + ((v < 4) ? (2 * v + p) : (16 + 2 * (v - 4) + p)); // g==0
      const int k1 = k0 + 8;                                                        // g==1
      const int ic0 = k0 / KK, r0 = k0 % KK, ky0 = r0 / KW, kx0 = r0 % KW;
      const int ic1 = k1 / KK, r1 = k1 % KK, ky1 = r1 / KW, kx1 = r1 % KW;
      const int c0 = (ic0 * HIN + ky0) * WIN + kx0;     // compile-time constants
      const int c1 = (ic1 * HIN + ky1) * WIN + kx1;
      a[e] = X[pixBase + (g ? c1 : c0)];
    }
    // ---- B fragments: one aligned 32B vector load each, then WMMA
#pragma unroll
    for (int t = 0; t < NT; ++t) {
      const v16h bf = fragv[(chunk * NT + t) * 32 + lane];
      acc[t] = __builtin_amdgcn_wmma_f32_16x16x32_f16(false, a, false, bf,
                                                      (short)0, acc[t], false, false);
    }
  }

  // ---- fused BN (eval) + ReLU epilogue; store into consumer's (padded) layout
#pragma unroll
  for (int t = 0; t < NT; ++t) {
    const int n  = t * 16 + mrow;
    const int nn = (n < OC) ? n : (OC - 1);             // clamped, branch-free
    const float gm = BN[0 * OC + nn], bt = BN[1 * OC + nn];
    const float mn = BN[2 * OC + nn], vr = BN[3 * OC + nn];
    const float s  = gm * rsqrtf(vr + BN_EPS);
    const float shift = bt - mn * s;
    const int rowBase = ((b * OC + nn) * OHP + oy + OPAD) * OWP + OPAD;
#pragma unroll
    for (int r = 0; r < 8; ++r) {
      const int xr = x0 + r + 8 * g;                    // D layout: VGPR r -> M=r+8*half
      if (wvalid && n < OC && xr < OW) {
        float o = acc[t][r] * s + shift;
        o = o > 0.f ? o : 0.f;
        Y[rowBase + xr] = (_Float16)o;
      }
    }
  }
}

// ------------------------------------------------------- 3x3 stride-2 maxpool
// Reads unpadded [B,C,H,W]; writes [B,C,OH+2*OP,OW+2*OP] interior at (OP,OP).
__global__ __launch_bounds__(256) void maxpool3s2_f16(
    const _Float16* __restrict__ X, _Float16* __restrict__ Y,
    int C, int H, int W, int OH, int OW, int OP, long total)
{
  long i = (long)blockIdx.x * blockDim.x + threadIdx.x;
  if (i >= total) return;
  int ox = (int)(i % OW); long t = i / OW;
  int oy = (int)(t % OH); t /= OH;
  int c  = (int)(t % C);
  int b  = (int)(t / C);
  const _Float16* base = X + (((long)b * C + c) * H + 2 * oy) * W + 2 * ox;
  float m = -1e30f;
#pragma unroll
  for (int ky = 0; ky < 3; ++ky)
#pragma unroll
    for (int kx = 0; kx < 3; ++kx)
      m = fmaxf(m, (float)base[ky * W + kx]);
  const int Hp = OH + 2 * OP, Wp = OW + 2 * OP;
  Y[(((long)b * C + c) * Hp + oy + OP) * Wp + ox + OP] = (_Float16)m;
}

// ------------------------------------------------- final FC: [512,64]@[64,183]
__global__ __launch_bounds__(256) void linear_wmma(
    const _Float16* __restrict__ H9, const _Float16* __restrict__ Lfrag,
    const float* __restrict__ bl, float* __restrict__ OUT)
{
  const int lane = threadIdx.x & 31;
  const int wvq  = threadIdx.x >> 5;
  const int mrow = lane & 15;
  const int g    = lane >> 4;
  const int tileN = blockIdx.x;             // 12 N tiles (183 -> 192)
  const int tileM = blockIdx.y * 8 + wvq;   // 32 M tiles (512)
  const int m0 = tileM * 16;
  const int n  = tileN * 16 + mrow;
  const v16h* __restrict__ fragv = (const v16h*)Lfrag;
  const _Float16* __restrict__ rowA = H9 + (long)(m0 + mrow) * 64 + 8 * g;

  v8f acc = {};
#pragma unroll
  for (int chunk = 0; chunk < 2; ++chunk) {
    v16h a;
#pragma unroll
    for (int e = 0; e < 16; ++e) {
      const int v = e >> 1, p = e & 1;
      const int kc = chunk * 32 + ((v < 4) ? (2 * v + p) : (16 + 2 * (v - 4) + p));
      a[e] = rowA[kc];
    }
    const v16h bf = fragv[(chunk * 12 + tileN) * 32 + lane];
    acc = __builtin_amdgcn_wmma_f32_16x16x32_f16(false, a, false, bf,
                                                 (short)0, acc, false, false);
  }
  if (n < 183) {
    const float bias = bl[n];
#pragma unroll
    for (int r = 0; r < 8; ++r) {
      const int m = m0 + r + 8 * g;
      OUT[(long)m * 183 + n] = acc[r] + bias;
    }
  }
}

// ---------------------------------------------------------------- dispatcher
extern "C" void kernel_launch(void* const* d_in, const int* in_sizes, int n_in,
                              void* d_out, int out_size, void* d_ws, size_t ws_size,
                              hipStream_t stream) {
  (void)in_sizes; (void)n_in; (void)out_size; (void)ws_size;
  const float* x = (const float*)d_in[0];
  const float* c[9]; const float* bn[9];
  for (int i = 0; i < 9; ++i) { c[i] = (const float*)d_in[1 + i]; bn[i] = (const float*)d_in[10 + i]; }
  const float* lw = (const float*)d_in[19];
  const float* lb = (const float*)d_in[20];
  float* out = (float*)d_out;

  char* ws = (char*)d_ws;
  const size_t HALF = 77074432;                 // 256-aligned; 4096 B less than 512*8*97*97*2
  _Float16* R0 = (_Float16*)ws;
  _Float16* R1 = (_Float16*)(ws + HALF);

  auto cdiv = [](long a, long b) { return (a + b - 1) / b; };

  // ---- weight fragment buffers
  static const int KTOTs[9] = {27, 72, 72, 144, 144, 216, 216, 288, 800};
  static const int OCs[9]   = {8, 8, 16, 16, 24, 24, 32, 32, 64};
  static const int CHs[9]   = {1, 3, 3, 5, 5, 7, 7, 9, 25};
  static const int NTs[9]   = {1, 1, 1, 1, 2, 2, 2, 2, 4};
  size_t off = 2 * HALF;
  _Float16* wf[9];
  for (int i = 0; i < 9; ++i) {
    wf[i] = (_Float16*)(ws + off);
    off += ((size_t)CHs[i] * NTs[i] * 1024 + 255) & ~(size_t)255;
  }
  _Float16* lf = (_Float16*)(ws + off);

  for (int i = 0; i < 9; ++i) {
    long tot = (long)CHs[i] * NTs[i] * 512;
    prep_wfrag<<<dim3((unsigned)cdiv(tot, 256)), 256, 0, stream>>>(
        c[i], wf[i], OCs[i], KTOTs[i], NTs[i], 0, 0, tot);
  }
  { long tot = 2L * 12 * 512;
    prep_wfrag<<<dim3((unsigned)cdiv(tot, 256)), 256, 0, stream>>>(
        lw, lf, 183, 64, 12, 1, 183, tot); }

#define BORDER(P, N, HP)                                                              \
  { long tot = (long)(N) * (2 * (HP) + 2 * ((HP) - 2));                               \
    zero_border<<<dim3((unsigned)cdiv(tot, 256)), 256, 0, stream>>>(P, HP, HP, tot); }
#define POOL(SRC, DST, C, H, OH, OP)                                                  \
  { long tot = 512L * (C) * (OH) * (OH);                                              \
    maxpool3s2_f16<<<dim3((unsigned)cdiv(tot, 256)), 256, 0, stream>>>(               \
        SRC, DST, C, H, H, OH, OH, OP, tot); }
#define CONV(IC, OC, KH, KW, PAD, H, W, OPAD, SRC, DST, Fp, BNp)                      \
  { const long OHc = (H) + 2 * (PAD) - (KH) + 1;                                      \
    const long OWc = (W) + 2 * (PAD) - (KW) + 1;                                      \
    long tiles = 512L * OHc * ((OWc + 15) / 16);                                      \
    conv_bnrelu_wmma<IC, OC, KH, KW, PAD, H, W, OPAD>                                 \
        <<<dim3((unsigned)cdiv(tiles, 8)), 256, 0, stream>>>(SRC, Fp, BNp, DST); }

  // ---- input f32 -> padded f16 in R1: [512,3,97,97]
  BORDER(R1, 512 * 3, 97);
  { long tot = 512L * 3 * 95 * 95;
    cvt_pad_f16<<<dim3((unsigned)cdiv(tot, 256)), 256, 0, stream>>>(x, R1, 3, 95, 95, tot); }

  BORDER(R0, 512 * 8, 97);
  CONV(3, 8, 3, 3, 1, 95, 95, 1, R1, R0, wf[0], bn[0]);     // -> R0 [8,97,97] padded
  CONV(8, 8, 3, 3, 1, 95, 95, 0, R0, R1, wf[1], bn[1]);     // -> R1 [8,95,95]
  BORDER(R0, 512 * 8, 49);
  POOL(R1, R0, 8, 95, 47, 1);                               // -> R0 [8,49,49] padded
  BORDER(R1, 512 * 16, 49);
  CONV(8, 16, 3, 3, 1, 47, 47, 1, R0, R1, wf[2], bn[2]);    // -> R1 [16,49,49] padded
  CONV(16, 16, 3, 3, 1, 47, 47, 0, R1, R0, wf[3], bn[3]);   // -> R0 [16,47,47]
  BORDER(R1, 512 * 16, 25);
  POOL(R0, R1, 16, 47, 23, 1);                              // -> R1 [16,25,25] padded
  BORDER(R0, 512 * 24, 25);
  CONV(16, 24, 3, 3, 1, 23, 23, 1, R1, R0, wf[4], bn[4]);   // -> R0 [24,25,25] padded
  CONV(24, 24, 3, 3, 1, 23, 23, 0, R0, R1, wf[5], bn[5]);   // -> R1 [24,23,23]
  BORDER(R0, 512 * 24, 13);
  POOL(R1, R0, 24, 23, 11, 1);                              // -> R0 [24,13,13] padded
  BORDER(R1, 512 * 32, 13);
  CONV(24, 32, 3, 3, 1, 11, 11, 1, R0, R1, wf[6], bn[6]);   // -> R1 [32,13,13] padded
  CONV(32, 32, 3, 3, 1, 11, 11, 0, R1, R0, wf[7], bn[7]);   // -> R0 [32,11,11]
  POOL(R0, R1, 32, 11, 5, 0);                               // -> R1 [32,5,5]
  CONV(32, 64, 5, 5, 0, 5, 5, 0, R1, R0, wf[8], bn[8]);     // -> R0 [512,64]

  linear_wmma<<<dim3(12, 4), 256, 0, stream>>>(R0, lf, lb, out);
#undef CONV
#undef POOL
#undef BORDER
}
